// PromptGenerator_1700807049316
// MI455X (gfx1250) — matrix operand order repaired and checked by reading, not compiled, not run on hardware
//
#include <hip/hip_runtime.h>

// ---------------------------------------------------------------------------
// MI455X / gfx1250 implementation. wave32, WMMA f16->f32.
// All GEMM-shaped work runs through v_wmma_f32_16x16x32_f16.
// ---------------------------------------------------------------------------

typedef __attribute__((ext_vector_type(16))) _Float16 v16h;
typedef __attribute__((ext_vector_type(8)))  float    v8f;
typedef _Float16 f16;

union U16x2 { unsigned u; f16 h[2]; };
union FragH { v16h v; unsigned u[8]; };

static __device__ __forceinline__ v8f vzero8() {
  v8f z;
#pragma unroll
  for (int i = 0; i < 8; i++) z[i] = 0.f;
  return z;
}

#define KPAD 1248   // 1225 keys zero-padded to a multiple of 32 (39*32)

// CDNA5 async global->LDS DMA path (ASYNCcnt-tracked). Probe round confirmed
// the builtin exists and takes pointers to 16-byte int vectors (AS1 src,
// AS3 dst). Require the asynccnt wait builtin too, else fall back to sync.
#if defined(__has_builtin)
#  if __has_builtin(__builtin_amdgcn_global_load_async_to_lds_b128) && \
      __has_builtin(__builtin_amdgcn_s_wait_asynccnt)
#    define HAVE_ASYNC_LDS 1
#  endif
#endif
#ifndef HAVE_ASYNC_LDS
#  define HAVE_ASYNC_LDS 0
#endif

#if HAVE_ASYNC_LDS
typedef int v4i_vs __attribute__((vector_size(16)));
typedef __attribute__((address_space(1))) v4i_vs g_v4i;   // global int4
typedef __attribute__((address_space(3))) v4i_vs l_v4i;   // LDS int4
#endif

// ---------------------------------------------------------------------------
// Generic batched GEMM:  out[b][m][n] = sum_k A[b][m][k] * Bw[b][n][k] (+bias)(+resid)
// REQUIRES: K % 32 == 0 and operand rows readable for K halves (pad with zeros).
// OOB slab rows are CLAMPED, not zeroed: a clamped A-row only affects discarded
// C-rows; a clamped B-row only affects discarded C-columns (always finite).
// Block tile 128(M) x 128(N), K-slab 32 staged in LDS. 8 waves, each 32x64.
// ---------------------------------------------------------------------------
__global__ __launch_bounds__(256) void gemm_wmma(
    const f16* __restrict__ A, const f16* __restrict__ Bw,
    const float* __restrict__ bias, const float* __restrict__ resid,
    float* __restrict__ outF, f16* __restrict__ outH,
    int M, int N, int K, int lda, int ldb, int ldo,
    long long sA, long long sB, long long sO)
{
  __shared__ f16 As[128 * 32];   // 8 KB
  __shared__ f16 Bs[128 * 32];   // 8 KB

  A  += (size_t)blockIdx.z * (size_t)sA;
  Bw += (size_t)blockIdx.z * (size_t)sB;
  const size_t ob = (size_t)blockIdx.z * (size_t)sO;

  const int m0 = blockIdx.y * 128, n0 = blockIdx.x * 128;
  const int tid = threadIdx.x, w = tid >> 5, l = tid & 31;
  const int wm = w & 3, wn = w >> 2;                  // wave tile: (wm*32, wn*64)

  v8f acc[2][4];
#pragma unroll
  for (int i = 0; i < 2; i++)
#pragma unroll
    for (int j = 0; j < 4; j++) acc[i][j] = vzero8();

  const int mrow = l & 15, koff = (l >> 4) * 8;       // A frag lane mapping
  const int nrow = l & 15, kb   = (l >> 4) * 16;      // B frag lane mapping

  for (int k0 = 0; k0 < K; k0 += 32) {
    // cooperative slab loads: each thread brings 2x8 halves per matrix (b128)
#pragma unroll
    for (int c = 0; c < 2; c++) {
      int idx  = tid * 2 + c;
      int row  = idx >> 2, kseg = (idx & 3) * 8;
      int kk   = k0 + kseg;                            // kk+8 <= K guaranteed
      int gr = m0 + row; if (gr >= M) gr = M - 1;      // clamp (see header note)
      int gn = n0 + row; if (gn >= N) gn = N - 1;
#if HAVE_ASYNC_LDS
      __builtin_amdgcn_global_load_async_to_lds_b128(
          (g_v4i*)(A + (size_t)gr * lda + kk), (l_v4i*)&As[row * 32 + kseg], 0, 0);
      __builtin_amdgcn_global_load_async_to_lds_b128(
          (g_v4i*)(Bw + (size_t)gn * ldb + kk), (l_v4i*)&Bs[row * 32 + kseg], 0, 0);
#else
      *(uint4*)&As[row * 32 + kseg] = *(const uint4*)(A + (size_t)gr * lda + kk);
      *(uint4*)&Bs[row * 32 + kseg] = *(const uint4*)(Bw + (size_t)gn * ldb + kk);
#endif
    }
    // speculative prefetch of the next K-slab (global_prefetch_b8)
    if (k0 + 32 < K) {
      int pm = m0 + (tid >> 1); if (pm >= M) pm = M - 1;
      int pn = n0 + (tid >> 1); if (pn >= N) pn = N - 1;
      __builtin_prefetch(A  + (size_t)pm * lda + k0 + 32 + (tid & 1) * 16, 0, 0);
      __builtin_prefetch(Bw + (size_t)pn * ldb + k0 + 32 + (tid & 1) * 16, 0, 0);
    }
#if HAVE_ASYNC_LDS
    __builtin_amdgcn_s_wait_asynccnt(0);
#endif
    __syncthreads();

    FragH fa[2], fb[4];
#pragma unroll
    for (int i = 0; i < 2; i++) {
      const f16* base = &As[(wm * 32 + i * 16 + mrow) * 32];
#pragma unroll
      for (int v = 0; v < 4; v++) fa[i].u[v]     = *(const unsigned*)(base + koff + 2 * v);
#pragma unroll
      for (int v = 0; v < 4; v++) fa[i].u[4 + v] = *(const unsigned*)(base + 16 + koff + 2 * v);
    }
#pragma unroll
    for (int j = 0; j < 4; j++) {
      const f16* base = &Bs[(wn * 64 + j * 16 + nrow) * 32];
#pragma unroll
      for (int v = 0; v < 4; v++) fb[j].u[v]     = *(const unsigned*)(base + koff + 2 * v);
#pragma unroll
      for (int v = 0; v < 4; v++) fb[j].u[4 + v] = *(const unsigned*)(base + 16 + koff + 2 * v);
    }
#pragma unroll
    for (int i = 0; i < 2; i++)
#pragma unroll
      for (int j = 0; j < 4; j++)
        acc[i][j] = __builtin_amdgcn_wmma_f32_16x16x32_f16(
            false, fa[i].v, false, fb[j].v, (short)0, acc[i][j], false, false);
    __syncthreads();
  }

  // epilogue: C/D layout lanes 0-15: n=lane, m=r; lanes 16-31: n=lane-16, m=r+8
#pragma unroll
  for (int i = 0; i < 2; i++) {
    int mb = m0 + wm * 32 + i * 16 + ((l >> 4) << 3);
#pragma unroll
    for (int j = 0; j < 4; j++) {
      int nn = n0 + wn * 64 + j * 16 + (l & 15);
      if (nn >= N) continue;
      float bv = bias ? bias[nn] : 0.f;
#pragma unroll
      for (int r = 0; r < 8; r++) {
        int mm = mb + r;
        if (mm >= M) continue;
        float v = acc[i][j][r] + bv;
        size_t oi = ob + (size_t)mm * ldo + nn;
        if (resid) v += resid[oi];
        if (outF) outF[oi] = v;
        if (outH) outH[oi] = (f16)v;
      }
    }
  }
}

// ---------------------------------------------------------------------------
// Fused self-attention: one block per (head, seq). qkv rows: [q(1024)|k|v].
// Dynamic LDS layout (172 KB, legal under gfx1250's 320 KB/WG):
//   sc  : f32[112][128]  raw scores / exp values
//   Q16 : f16[112][128]  Q staged row-major (rows clamped; garbage rows unused)
//   K16 : f16[112][128]  K staged row-major (clamped; cols >= S masked by P16)
//   P16 : f16[112][128]  normalized probabilities (zeroed: the real mask)
//   VT  : f16[128][128]  V transposed (clamped keys; masked by P16 zeros)
// Both WMMA phases feed exclusively from LDS (no predicated VMEM).
// ---------------------------------------------------------------------------
#define ATTN_LDS_BYTES (57344 + 28672 + 28672 + 28672 + 32768)   // 176128 B

__global__ __launch_bounds__(256) void attn_self(
    const f16* __restrict__ qkv, f16* __restrict__ outH, int S)
{
  extern __shared__ char smem[];
  float* sc  = (float*)smem;                       // 57344 B
  f16*  Q16  = (f16*)(smem + 57344);               // 28672 B
  f16*  K16s = (f16*)(smem + 86016);               // 28672 B
  f16*  P16  = (f16*)(smem + 114688);              // 28672 B
  f16*  VT   = (f16*)(smem + 143360);              // 32768 B

  const int hd = blockIdx.x;                       // head
  const int seq = blockIdx.y;
  const size_t base = (size_t)seq * S * 3072;
  const int nt = (S + 15) >> 4;
  const int tid = threadIdx.x, w = tid >> 5, l = tid & 31;
  const float scale = 0.08838834764831845f;        // 1/sqrt(128)

  // ---- stage Q,K row-major and V transposed (clamped rows); zero P16 ----
  for (int i = tid; i < 112 * 16; i += 256) {      // tok (112) x 16 segments of 8
    int tok = i >> 4, seg = (i & 15) * 8;
    int tc = (tok < S) ? tok : S - 1;
    const f16* p = qkv + base + (size_t)tc * 3072 + hd * 128 + seg;
    *(uint4*)&Q16[tok * 128 + seg]  = *(const uint4*)p;
    *(uint4*)&K16s[tok * 128 + seg] = *(const uint4*)(p + 1024);
  }
  for (int i = tid; i < 128 * 16; i += 256) {      // kk (128) x 16 segments of 8 d's
    int kk = i >> 4, seg = (i & 15) * 8;
    int kc = (kk < S) ? kk : S - 1;
    uint4 vv = *(const uint4*)(qkv + base + (size_t)kc * 3072 + 2048 + hd * 128 + seg);
    alignas(16) f16 tmp[8];
    *(uint4*)tmp = vv;
#pragma unroll
    for (int e = 0; e < 8; e++) VT[(seg + e) * 128 + kk] = tmp[e];
  }
  {
    unsigned* p16u = (unsigned*)P16;
    for (int i = tid; i < 112 * 128 / 2; i += 256) p16u[i] = 0u;
  }
  __syncthreads();

  const int mrow = l & 15, koff = (l >> 4) * 8, kb = (l >> 4) * 16;

  // ---- scores = (Q K^T) * scale : pure LDS fragments ----
  for (int t = w; t < nt * nt; t += 8) {
    int mt = t / nt, nc = t % nt;
    v8f acc = vzero8();
    const f16* qrow = Q16  + (mt * 16 + mrow) * 128;
    const f16* krow = K16s + (nc * 16 + (l & 15)) * 128;
#pragma unroll
    for (int kc = 0; kc < 4; kc++) {
      FragH fa, fb;
#pragma unroll
      for (int v = 0; v < 4; v++) fa.u[v]     = *(const unsigned*)(qrow + kc * 32 + koff + 2 * v);
#pragma unroll
      for (int v = 0; v < 4; v++) fa.u[4 + v] = *(const unsigned*)(qrow + kc * 32 + 16 + koff + 2 * v);
#pragma unroll
      for (int v = 0; v < 4; v++) fb.u[v]     = *(const unsigned*)(krow + kc * 32 + koff + 2 * v);
#pragma unroll
      for (int v = 0; v < 4; v++) fb.u[4 + v] = *(const unsigned*)(krow + kc * 32 + 16 + koff + 2 * v);
      acc = __builtin_amdgcn_wmma_f32_16x16x32_f16(
          false, fa.v, false, fb.v, (short)0, acc, false, false);
    }
    int nn = nc * 16 + (l & 15);
    int mb = mt * 16 + ((l >> 4) << 3);
#pragma unroll
    for (int r = 0; r < 8; r++) sc[(mb + r) * 128 + nn] = acc[r] * scale;
  }
  __syncthreads();

  // ---- row softmax (wave per row) -> packed f16 probabilities ----
  for (int r = w; r < S; r += 8) {
    float* row = sc + r * 128;
    f16* prow = P16 + r * 128;
    float mx = -1e30f;
    for (int c = l; c < S; c += 32) mx = fmaxf(mx, row[c]);
    for (int o = 16; o > 0; o >>= 1) mx = fmaxf(mx, __shfl_xor(mx, o, 32));
    float sum = 0.f;
    for (int c = l; c < S; c += 32) { float p = __expf(row[c] - mx); row[c] = p; sum += p; }
    for (int o = 16; o > 0; o >>= 1) sum += __shfl_xor(sum, o, 32);
    float inv = 1.f / sum;
    for (int c = l; c < 128; c += 32) prow[c] = (c < S) ? (f16)(row[c] * inv) : (f16)0.f;
  }
  __syncthreads();

  // ---- O = P V  (A from P16, B from VT; pure LDS pair loads) ----
  const int kcn = (nt * 16 + 31) >> 5;               // 4 for S=98, 2 for S=49
  for (int t = w; t < nt * 8; t += 8) {
    int mt = t >> 3, nc = t & 7;
    v8f acc = vzero8();
    const f16* prow  = P16 + (mt * 16 + mrow) * 128;
    const f16* vtrow = VT + (nc * 16 + (l & 15)) * 128;
    for (int kc = 0; kc < kcn; kc++) {
      FragH fa, fb;
#pragma unroll
      for (int v = 0; v < 4; v++) fa.u[v]     = *(const unsigned*)(prow + kc * 32 + koff + 2 * v);
#pragma unroll
      for (int v = 0; v < 4; v++) fa.u[4 + v] = *(const unsigned*)(prow + kc * 32 + 16 + koff + 2 * v);
#pragma unroll
      for (int v = 0; v < 4; v++) fb.u[v]     = *(const unsigned*)(vtrow + kc * 32 + koff + 2 * v);
#pragma unroll
      for (int v = 0; v < 4; v++) fb.u[4 + v] = *(const unsigned*)(vtrow + kc * 32 + 16 + koff + 2 * v);
      acc = __builtin_amdgcn_wmma_f32_16x16x32_f16(
          false, fa.v, false, fb.v, (short)0, acc, false, false);
    }
    int nn = hd * 128 + nc * 16 + (l & 15);
    int mb = mt * 16 + ((l >> 4) << 3);
#pragma unroll
    for (int r = 0; r < 8; r++) {
      int tok = mb + r;
      if (tok < S) outH[((size_t)seq * S + tok) * 1024 + nn] = (f16)acc[r];
    }
  }
}

// ---------------------------------------------------------------------------
// Elementwise / reduction helpers
// ---------------------------------------------------------------------------
__global__ void cvt_f32_to_f16(const float* __restrict__ s, f16* __restrict__ d, int n)
{
  int i = blockIdx.x * 256 + threadIdx.x;
  if (i < n) d[i] = (f16)s[i];
}

__global__ __launch_bounds__(256) void ln_rows(
    const float* __restrict__ x, const float* __restrict__ gw,
    const float* __restrict__ gb, f16* __restrict__ out)
{
  const int row = blockIdx.x;
  const float* xr = x + (size_t)row * 1024;
  __shared__ float red[16];
  const int tid = threadIdx.x, wv = tid >> 5, l = tid & 31;
  float v[4], s = 0.f, s2 = 0.f;
#pragma unroll
  for (int i = 0; i < 4; i++) { v[i] = xr[tid + 256 * i]; s += v[i]; s2 += v[i] * v[i]; }
  for (int o = 16; o > 0; o >>= 1) { s += __shfl_xor(s, o, 32); s2 += __shfl_xor(s2, o, 32); }
  if (l == 0) { red[wv] = s; red[8 + wv] = s2; }
  __syncthreads();
  if (tid == 0) {
    float a = 0.f, c = 0.f;
    for (int i = 0; i < 8; i++) { a += red[i]; c += red[8 + i]; }
    red[0] = a; red[1] = c;
  }
  __syncthreads();
  float mean = red[0] * (1.f / 1024.f);
  float var  = red[1] * (1.f / 1024.f) - mean * mean;
  float rstd = rsqrtf(var + 1e-5f);
#pragma unroll
  for (int i = 0; i < 4; i++) {
    int c = tid + 256 * i;
    out[(size_t)row * 1024 + c] = (f16)(((v[i] - mean) * rstd) * gw[c] + gb[c]);
  }
}

__global__ void gauss49(float* __restrict__ g)
{
  int q = threadIdx.x;
  if (q >= 49) return;
  float cy = (float)(q / 7), cx = (float)(q % 7);
  float tmp[49], sum = 0.f;
  for (int i = 0; i < 7; i++)
    for (int j = 0; j < 7; j++) {
      float d = (i - cy) * (i - cy) + (j - cx) * (j - cx);
      float e = __expf(-0.5f * d);
      tmp[i * 7 + j] = e; sum += e;
    }
  float inv = 1.f / sum;
  for (int k = 0; k < 49; k++) g[q * 49 + k] = tmp[k] * inv;
}

__global__ void gather_qimg(const float* __restrict__ qf, float* __restrict__ out)
{
  size_t i = (size_t)blockIdx.x * 256 + threadIdx.x;
  if (i >= (size_t)1568 * 1024) return;
  int d = (int)(i & 1023);
  size_t r = i >> 10;
  int b = (int)(r / 49), p = (int)(r % 49);
  int ii = p / 7, jj = p % 7;
  out[i] = qf[((size_t)b * 98 + ii * 14 + jj) * 1024 + d];
}

__global__ void repack_s(const float* __restrict__ sup, f16* __restrict__ img, f16* __restrict__ msk)
{
  size_t i = (size_t)blockIdx.x * 256 + threadIdx.x;
  if (i >= (size_t)39200 * 1024) return;
  int d = (int)(i & 1023);
  size_t r = i >> 10;
  int bn = (int)(r / 49), p = (int)(r % 49);
  int ii = p / 7, jj = p % 7;
  img[i] = (f16)sup[((size_t)bn * 98 + ii * 14 + jj) * 1024 + d];
  msk[i] = (f16)sup[((size_t)bn * 98 + ii * 14 + 7 + jj) * 1024 + d];
}

__global__ __launch_bounds__(256) void softmax_gauss(
    const float* __restrict__ sc, const float* __restrict__ g, f16* __restrict__ attn)
{
  const int row = blockIdx.x;                 // b*49 + q
  const int q = row % 49;
  const float* sr = sc + (size_t)row * 1232;
  f16* ar = attn + (size_t)row * KPAD;
  __shared__ float red[8];
  const int tid = threadIdx.x, wv = tid >> 5, l = tid & 31;
  float vals[5];
  float mx = -1e30f;
#pragma unroll
  for (int i = 0; i < 5; i++) {
    int c = tid + 256 * i;
    float v = -1e30f;
    if (c < 1225) v = sr[c] * g[q * 49 + (c % 49)];
    vals[i] = v;
    mx = fmaxf(mx, v);
  }
  for (int o = 16; o > 0; o >>= 1) mx = fmaxf(mx, __shfl_xor(mx, o, 32));
  if (l == 0) red[wv] = mx;
  __syncthreads();
  float M = red[0];
  for (int i = 1; i < 8; i++) M = fmaxf(M, red[i]);
  __syncthreads();
  float sum = 0.f;
#pragma unroll
  for (int i = 0; i < 5; i++) {
    int c = tid + 256 * i;
    if (c < 1225) { vals[i] = __expf(vals[i] - M); sum += vals[i]; }
  }
  for (int o = 16; o > 0; o >>= 1) sum += __shfl_xor(sum, o, 32);
  if (l == 0) red[wv] = sum;
  __syncthreads();
  float S = 0.f;
  for (int i = 0; i < 8; i++) S += red[i];
  float inv = 1.f / S;
#pragma unroll
  for (int i = 0; i < 5; i++) {
    int c = tid + 256 * i;
    if (c < KPAD) ar[c] = (c < 1225) ? (f16)(vals[i] * inv) : (f16)0.f;
  }
}

__global__ void transpose_kv(const f16* __restrict__ src, f16* __restrict__ dst)
{
  // src: [32*1225][1024]  ->  dst: [32][1024][KPAD] (zero-padded tail)
  size_t i = (size_t)blockIdx.x * 256 + threadIdx.x;
  if (i >= (size_t)32 * 1024 * KPAD) return;
  int kk = (int)(i % KPAD);
  size_t r = i / KPAD;
  int d = (int)(r % 1024), b = (int)(r / 1024);
  dst[i] = (kk < 1225) ? src[((size_t)b * 1225 + kk) * 1024 + d] : (f16)0.f;
}

__global__ void assemble_out(
    const float* __restrict__ o1, const float* __restrict__ o2,
    const float* __restrict__ qimg2, const float* __restrict__ qf,
    float* __restrict__ out)
{
  size_t i = (size_t)blockIdx.x * 256 + threadIdx.x;
  if (i >= (size_t)32 * 196 * 1024) return;
  int d = (int)(i & 1023);
  size_t r = i >> 10;
  int tok = (int)(r % 196), b = (int)(r / 196);
  int r14 = tok / 14, c14 = tok % 14;
  float v;
  if (r14 < 7) {
    if (c14 < 7) v = o1[((size_t)b * 49 + r14 * 7 + c14) * 1024 + d];
    else         v = o2[((size_t)b * 49 + r14 * 7 + (c14 - 7)) * 1024 + d];
  } else {
    int rr = r14 - 7;
    if (c14 < 7) v = qimg2[((size_t)b * 49 + rr * 7 + c14) * 1024 + d];
    else         v = qf[((size_t)b * 98 + rr * 14 + c14) * 1024 + d];
  }
  out[i] = v;
}

// ---------------------------------------------------------------------------
// Host-side orchestration
// ---------------------------------------------------------------------------
extern "C" void kernel_launch(void* const* d_in, const int* in_sizes, int n_in,
                              void* d_out, int out_size, void* d_ws, size_t ws_size,
                              hipStream_t stream)
{
  (void)in_sizes; (void)n_in; (void)out_size; (void)ws_size;
  const float* supIn  = (const float*)d_in[0];
  const float* qfIn   = (const float*)d_in[1];
  const float* ln_w   = (const float*)d_in[2];
  const float* ln_b   = (const float*)d_in[3];
  const float* s_in_w = (const float*)d_in[4];
  const float* s_in_b = (const float*)d_in[5];
  const float* s_out_w= (const float*)d_in[6];
  const float* s_out_b= (const float*)d_in[7];
  const float* q_in_w = (const float*)d_in[8];
  const float* q_in_b = (const float*)d_in[9];
  const float* q_out_w= (const float*)d_in[10];
  const float* q_out_b= (const float*)d_in[11];
  const float* Wq = (const float*)d_in[12]; const float* bq  = (const float*)d_in[13];
  const float* Wk = (const float*)d_in[14]; const float* bk  = (const float*)d_in[15];
  const float* Wv1= (const float*)d_in[16]; const float* bv1 = (const float*)d_in[17];
  const float* Wv2= (const float*)d_in[18]; const float* bv2 = (const float*)d_in[19];

  const size_t SUPR = 78400;   // 32*25*98
  const size_t QIR  = 1568;    // 32*49
  const size_t SKVR = 39200;   // 32*25*49

  char* ws = (char*)d_ws;
  size_t off = 0;
  auto al = [&](size_t b) { size_t r = off; off = (off + b + 255) & ~(size_t)255; return r; };

  // persistent region (~390 MB)
  size_t o_sin   = al((size_t)3072 * 1024 * 2);
  size_t o_sout  = al((size_t)1024 * 1024 * 2);
  size_t o_qin   = al((size_t)3072 * 1024 * 2);
  size_t o_qout  = al((size_t)1024 * 1024 * 2);
  size_t o_wq    = al((size_t)1024 * 1024 * 2);
  size_t o_wk    = al((size_t)1024 * 1024 * 2);
  size_t o_wv1   = al((size_t)1024 * 1024 * 2);
  size_t o_wv2   = al((size_t)1024 * 1024 * 2);
  size_t o_g     = al((size_t)49 * 49 * 4);
  size_t o_sup   = al(SUPR * 1024 * 4);
  size_t o_qimg  = al(QIR * 1024 * 4);
  size_t o_qimg2f= al(QIR * 1024 * 4);
  size_t o_qimg2h= al(QIR * 1024 * 2);
  size_t o_o1    = al(QIR * 1024 * 4);
  size_t o_o2    = al(QIR * 1024 * 4);
  size_t o_qp16  = al(QIR * 1024 * 2);
  size_t o_score = al((size_t)32 * 49 * 1232 * 4);
  size_t o_attnp = al((size_t)32 * 49 * KPAD * 2);

  // phase-reused arena (~819 MB): phase 1 = MHA buffers, phase 2 = cross-attn
  size_t o_big    = off;
  size_t o_ln16   = o_big;
  size_t o_qkv16  = o_ln16  + SUPR * 1024 * 2;
  size_t o_attn16 = o_qkv16 + SUPR * 3072 * 2;
  size_t o_qln16  = o_attn16 + SUPR * 1024 * 2;
  size_t o_qqkv16 = o_qln16  + QIR * 1024 * 2;
  size_t o_qattn16= o_qqkv16 + QIR * 3072 * 2;
  // phase 2 aliases (phase 1 buffers are dead before these are written)
  size_t o_simg  = o_big;
  size_t o_smask = o_simg  + SKVR * 1024 * 2;
  size_t o_k16   = o_smask + SKVR * 1024 * 2;
  size_t o_v1    = o_k16   + SKVR * 1024 * 2;
  size_t o_v2    = o_v1    + SKVR * 1024 * 2;
  size_t o_v1T   = o_v2    + SKVR * 1024 * 2;
  size_t o_v2T   = o_v1T   + (size_t)32 * 1024 * KPAD * 2;

#define WP(T, o) ((T*)(ws + (o)))
  dim3 blk(256);
  const int NW = (3072 * 1024 + 255) / 256, NE = (1024 * 1024 + 255) / 256;

  // 0. weights -> f16, gaussian bank
  cvt_f32_to_f16<<<NW, blk, 0, stream>>>(s_in_w,  WP(f16, o_sin),  3072 * 1024);
  cvt_f32_to_f16<<<NE, blk, 0, stream>>>(s_out_w, WP(f16, o_sout), 1024 * 1024);
  cvt_f32_to_f16<<<NW, blk, 0, stream>>>(q_in_w,  WP(f16, o_qin),  3072 * 1024);
  cvt_f32_to_f16<<<NE, blk, 0, stream>>>(q_out_w, WP(f16, o_qout), 1024 * 1024);
  cvt_f32_to_f16<<<NE, blk, 0, stream>>>(Wq,  WP(f16, o_wq),  1024 * 1024);
  cvt_f32_to_f16<<<NE, blk, 0, stream>>>(Wk,  WP(f16, o_wk),  1024 * 1024);
  cvt_f32_to_f16<<<NE, blk, 0, stream>>>(Wv1, WP(f16, o_wv1), 1024 * 1024);
  cvt_f32_to_f16<<<NE, blk, 0, stream>>>(Wv2, WP(f16, o_wv2), 1024 * 1024);
  gauss49<<<1, 64, 0, stream>>>(WP(float, o_g));

  // 1. support branch MHA: sup = qss + OutProj(Attn(QKV(LN(qss))))
  ln_rows<<<(int)SUPR, blk, 0, stream>>>(supIn, ln_w, ln_b, WP(f16, o_ln16));
  gemm_wmma<<<dim3(24, 613, 1), blk, 0, stream>>>(
      WP(f16, o_ln16), WP(f16, o_sin), s_in_b, nullptr, nullptr, WP(f16, o_qkv16),
      78400, 3072, 1024, 1024, 1024, 3072, 0, 0, 0);
  attn_self<<<dim3(8, 800), blk, ATTN_LDS_BYTES, stream>>>(
      WP(f16, o_qkv16), WP(f16, o_attn16), 98);
  gemm_wmma<<<dim3(8, 613, 1), blk, 0, stream>>>(
      WP(f16, o_attn16), WP(f16, o_sout), s_out_b, supIn, WP(float, o_sup), nullptr,
      78400, 1024, 1024, 1024, 1024, 1024, 0, 0, 0);

  // 2. query branch MHA on the 7x7 image half
  gather_qimg<<<(int)((QIR * 1024 + 255) / 256), blk, 0, stream>>>(qfIn, WP(float, o_qimg));
  ln_rows<<<(int)QIR, blk, 0, stream>>>(WP(float, o_qimg), ln_w, ln_b, WP(f16, o_qln16));
  gemm_wmma<<<dim3(24, 13, 1), blk, 0, stream>>>(
      WP(f16, o_qln16), WP(f16, o_qin), q_in_b, nullptr, nullptr, WP(f16, o_qqkv16),
      1568, 3072, 1024, 1024, 1024, 3072, 0, 0, 0);
  attn_self<<<dim3(8, 32), blk, ATTN_LDS_BYTES, stream>>>(
      WP(f16, o_qqkv16), WP(f16, o_qattn16), 49);
  gemm_wmma<<<dim3(8, 13, 1), blk, 0, stream>>>(
      WP(f16, o_qattn16), WP(f16, o_qout), q_out_b, WP(float, o_qimg),
      WP(float, o_qimg2f), WP(f16, o_qimg2h),
      1568, 1024, 1024, 1024, 1024, 1024, 0, 0, 0);

  // 3. cross-attention projections (arena reused)
  repack_s<<<(int)((SKVR * 1024 + 255) / 256), blk, 0, stream>>>(
      WP(float, o_sup), WP(f16, o_simg), WP(f16, o_smask));
  gemm_wmma<<<dim3(8, 307, 1), blk, 0, stream>>>(
      WP(f16, o_simg), WP(f16, o_wk), bk, nullptr, nullptr, WP(f16, o_k16),
      39200, 1024, 1024, 1024, 1024, 1024, 0, 0, 0);
  gemm_wmma<<<dim3(8, 307, 1), blk, 0, stream>>>(
      WP(f16, o_simg), WP(f16, o_wv1), bv1, nullptr, nullptr, WP(f16, o_v1),
      39200, 1024, 1024, 1024, 1024, 1024, 0, 0, 0);
  gemm_wmma<<<dim3(8, 307, 1), blk, 0, stream>>>(
      WP(f16, o_smask), WP(f16, o_wv2), bv2, nullptr, nullptr, WP(f16, o_v2),
      39200, 1024, 1024, 1024, 1024, 1024, 0, 0, 0);
  gemm_wmma<<<dim3(8, 13, 1), blk, 0, stream>>>(
      WP(f16, o_qimg2h), WP(f16, o_wq), bq, nullptr, nullptr, WP(f16, o_qp16),
      1568, 1024, 1024, 1024, 1024, 1024, 0, 0, 0);
  transpose_kv<<<(int)(((size_t)32 * 1024 * KPAD + 255) / 256), blk, 0, stream>>>(
      WP(f16, o_v1), WP(f16, o_v1T));
  transpose_kv<<<(int)(((size_t)32 * 1024 * KPAD + 255) / 256), blk, 0, stream>>>(
      WP(f16, o_v2), WP(f16, o_v2T));

  // 4. score GEMM (batched over B=32), gaussian-weighted softmax, P*V GEMMs
  gemm_wmma<<<dim3(10, 1, 32), blk, 0, stream>>>(
      WP(f16, o_qp16), WP(f16, o_k16), nullptr, nullptr, WP(float, o_score), nullptr,
      49, 1225, 1024, 1024, 1024, 1232,
      (long long)49 * 1024, (long long)1225 * 1024, (long long)49 * 1232);
  softmax_gauss<<<1568, blk, 0, stream>>>(WP(float, o_score), WP(float, o_g), WP(f16, o_attnp));
  gemm_wmma<<<dim3(8, 1, 32), blk, 0, stream>>>(
      WP(f16, o_attnp), WP(f16, o_v1T), nullptr, nullptr, WP(float, o_o1), nullptr,
      49, 1024, KPAD, KPAD, KPAD, 1024,
      (long long)49 * KPAD, (long long)1024 * KPAD, (long long)49 * 1024);
  gemm_wmma<<<dim3(8, 1, 32), blk, 0, stream>>>(
      WP(f16, o_attnp), WP(f16, o_v2T), nullptr, nullptr, WP(float, o_o2), nullptr,
      49, 1024, KPAD, KPAD, KPAD, 1024,
      (long long)49 * KPAD, (long long)1024 * KPAD, (long long)49 * 1024);

  // 5. assemble (B,196,E)
  assemble_out<<<(int)(((size_t)32 * 196 * 1024 + 255) / 256), blk, 0, stream>>>(
      WP(float, o_o1), WP(float, o_o2), WP(float, o_qimg2f), qfIn, (float*)d_out);
#undef WP
}
